// SAEStandard_35579509080449
// MI455X (gfx1250) — matrix operand order repaired and checked by reading, not compiled, not run on hardware
//
#include <hip/hip_runtime.h>
#include <hip/hip_bf16.h>
#include <cstdint>

typedef __attribute__((ext_vector_type(16))) __bf16        v16bf;
typedef __attribute__((ext_vector_type(8)))  float         v8f;
typedef __attribute__((ext_vector_type(4)))  unsigned int  u32x4;

#define DIMIN      768
#define WIDTH      12288
#define NROWS      16384
#define TOPK       32

__device__ __forceinline__ unsigned short f32_to_bf16_rne(float f) {
    unsigned u = __float_as_uint(f);
    unsigned r = u + 0x7FFFu + ((u >> 16) & 1u);   // round-to-nearest-even
    return (unsigned short)(r >> 16);
}

union FragBF {
    v16bf          v;
    unsigned short h[16];
    u32x4          q[2];
};

// A fragment (16-bit, 16x32): lane<16 halves = K kk..kk+15 contiguous,
//                             lane>=16      = K kk+16..kk+31 contiguous
// B fragment (16-bit, 32x16): lane<16 halves = K kk..kk+15 contiguous,
//                             lane>=16      = K kk+16..kk+31 contiguous
__device__ __forceinline__ void load_frag_a(FragBF& a, const unsigned short* row,
                                            int kk, int hi) {
    a.q[0] = *(const u32x4*)&row[kk + hi * 16];
    a.q[1] = *(const u32x4*)&row[kk + hi * 16 + 8];
}
__device__ __forceinline__ void load_frag_b(FragBF& b, const unsigned short* row,
                                            int kk, int hi) {
    b.q[0] = *(const u32x4*)&row[kk + hi * 16];
    b.q[1] = *(const u32x4*)&row[kk + hi * 16 + 8];
}

// ---------------------------------------------------------------------------
// Prep kernels: Ae f32 -> bf16 ; xc = (x - bd) f32 -> bf16
// ---------------------------------------------------------------------------
__global__ void cvt_bf16_kernel(const float* __restrict__ src,
                                unsigned short* __restrict__ dst, int n) {
    int i = blockIdx.x * blockDim.x + threadIdx.x;
    int stride = gridDim.x * blockDim.x;
    for (int j = i; j < n; j += stride) dst[j] = f32_to_bf16_rne(src[j]);
}

__global__ void prep_xc_kernel(const float* __restrict__ x,
                               const float* __restrict__ bd,
                               unsigned short* __restrict__ dst, int n) {
    int i = blockIdx.x * blockDim.x + threadIdx.x;
    int stride = gridDim.x * blockDim.x;
    for (int j = i; j < n; j += stride) {
        int c = j % DIMIN;
        dst[j] = f32_to_bf16_rne(x[j] - bd[c]);
    }
}

// ---------------------------------------------------------------------------
// Main kernel (workspace path): 32 rows per WG, 8 waves.
// Per 128-column chunk each wave computes one 16-col B fragment stream and
// TWO row-tiles (B reused by 2 WMMAs). Streaming top-32 + sparse decode fused.
// ---------------------------------------------------------------------------
#define W_TILE_ROWS 32
#define W_CHUNK     128                       // 8 waves * 16 cols
#define W_NCHUNKS   (WIDTH / W_CHUNK)         // 96
#define W_CAP       128                       // max candidates/row/chunk (exact)

__global__ __launch_bounds__(256)
void sae_main_ws(const float* __restrict__ Ae,            // f32, decode gathers
                 const unsigned short* __restrict__ Aeb,  // bf16 Ae
                 const unsigned short* __restrict__ Xcb,  // bf16 (x - bd)
                 const float* __restrict__ bd,
                 const float* __restrict__ lambda_pre,
                 float* __restrict__ out) {
    __shared__ float          candV[W_TILE_ROWS][W_CAP];   // 16 KB
    __shared__ unsigned short candI[W_TILE_ROWS][W_CAP];   // 8 KB
    __shared__ unsigned int   cnt[W_TILE_ROWS];
    __shared__ float          topV[W_TILE_ROWS][TOPK];     // 4 KB
    __shared__ unsigned short topI[W_TILE_ROWS][TOPK];     // 2 KB
    __shared__ float          rowmin[W_TILE_ROWS];
    __shared__ float          bdS[DIMIN];                  // 3 KB

    const int tid     = threadIdx.x;
    const int lane    = tid & 31;
    const int wid     = tid >> 5;
    const int rowBase = blockIdx.x * W_TILE_ROWS;
    const int hi      = (lane >> 4) & 1;
    const int arow    = lane & 15;

    for (int i = tid; i < DIMIN; i += 256) bdS[i] = bd[i];
    if (tid < W_TILE_ROWS) { cnt[tid] = 0u; rowmin[tid] = 0.0f; }
    {
        float* tv = &topV[0][0];
        unsigned short* ti = &topI[0][0];
        for (int i = tid; i < W_TILE_ROWS * TOPK; i += 256) { tv[i] = 0.0f; ti[i] = 0; }
    }
    __syncthreads();

    const unsigned short* A0 = &Xcb[(size_t)(rowBase + arow) * DIMIN];
    const unsigned short* A1 = &Xcb[(size_t)(rowBase + 16 + arow) * DIMIN];

    for (int ch = 0; ch < W_NCHUNKS; ++ch) {
        const int colBase = ch * W_CHUNK + wid * 16;
        const unsigned short* Bp = &Aeb[(size_t)(colBase + arow) * DIMIN];
        if (ch + 1 < W_NCHUNKS)
            __builtin_prefetch(Bp + (size_t)W_CHUNK * DIMIN, 0, 1);

        v8f acc0 = {0.f, 0.f, 0.f, 0.f, 0.f, 0.f, 0.f, 0.f};
        v8f acc1 = {0.f, 0.f, 0.f, 0.f, 0.f, 0.f, 0.f, 0.f};

        #pragma unroll 4
        for (int kk = 0; kk < DIMIN; kk += 32) {
            FragBF a0, a1, b;
            load_frag_a(a0, A0, kk, hi);
            load_frag_a(a1, A1, kk, hi);
            load_frag_b(b, Bp, kk, hi);
            acc0 = __builtin_amdgcn_wmma_f32_16x16x32_bf16(
                       false, a0.v, false, b.v, (short)0, acc0, false, false);
            acc1 = __builtin_amdgcn_wmma_f32_16x16x32_bf16(
                       false, a1.v, false, b.v, (short)0, acc1, false, false);
        }

        const int col = colBase + (lane & 15);
        #pragma unroll
        for (int r = 0; r < 8; ++r) {
            const int m0 = r + (hi << 3);          // rows 0..15  (tile 0)
            const int m1 = m0 + 16;                // rows 16..31 (tile 1)
            float v0 = acc0[r];
            if (v0 > 0.0f && v0 > rowmin[m0]) {
                unsigned p = atomicAdd(&cnt[m0], 1u);
                if (p < W_CAP) { candV[m0][p] = v0; candI[m0][p] = (unsigned short)col; }
            }
            float v1 = acc1[r];
            if (v1 > 0.0f && v1 > rowmin[m1]) {
                unsigned p = atomicAdd(&cnt[m1], 1u);
                if (p < W_CAP) { candV[m1][p] = v1; candI[m1][p] = (unsigned short)col; }
            }
        }
        __syncthreads();

        if (tid < W_TILE_ROWS) {                   // per-row sorted-insert drain
            const int m = tid;
            unsigned n = cnt[m]; if (n > W_CAP) n = W_CAP;
            for (unsigned i = 0; i < n; ++i) {
                float v = candV[m][i];
                if (v > topV[m][TOPK - 1]) {
                    int j = TOPK - 1;
                    while (j > 0 && topV[m][j - 1] < v) {
                        topV[m][j] = topV[m][j - 1];
                        topI[m][j] = topI[m][j - 1];
                        --j;
                    }
                    topV[m][j] = v;
                    topI[m][j] = candI[m][i];
                }
            }
            rowmin[m] = topV[m][TOPK - 1];
            cnt[m] = 0u;
        }
        __syncthreads();
    }

    // sparse decode: out[r] = lam * sum_k val_k * Ae[idx_k,:] + bd
    float lam;
    {
        float lp = lambda_pre[0];
        lam = (lp > 20.0f) ? lp : log1pf(expf(lp));
    }
    for (int r = 0; r < W_TILE_ROWS; ++r) {
        for (int cb = 0; cb < DIMIN; cb += 256) {
            const int c = cb + tid;
            float acc = bdS[c];
            #pragma unroll 8
            for (int k = 0; k < TOPK; ++k) {
                float v = topV[r][k];
                int   w = (int)topI[r][k];
                acc += lam * v * Ae[(size_t)w * DIMIN + c];
            }
            out[(size_t)(rowBase + r) * DIMIN + c] = acc;
        }
    }
}

// ---------------------------------------------------------------------------
// Fallback (no workspace): 16 rows per WG, xc tile in LDS, inline f32->bf16
// conversion of Ae fragments. Same top-k + decode structure.
// ---------------------------------------------------------------------------
#define F_TILE_ROWS 16
#define F_CHUNK     256
#define F_NCHUNKS   (WIDTH / F_CHUNK)
#define F_CAP       256

__global__ __launch_bounds__(256)
void sae_main_nows(const float* __restrict__ x,
                   const float* __restrict__ Ae,
                   const float* __restrict__ bd,
                   const float* __restrict__ lambda_pre,
                   float* __restrict__ out) {
    __shared__ unsigned short xcb[F_TILE_ROWS * DIMIN];
    __shared__ float bdS[DIMIN];
    __shared__ float          candV[F_TILE_ROWS][F_CAP];
    __shared__ unsigned short candI[F_TILE_ROWS][F_CAP];
    __shared__ unsigned int   cnt[F_TILE_ROWS];
    __shared__ float topV[F_TILE_ROWS][TOPK];
    __shared__ int   topI[F_TILE_ROWS][TOPK];
    __shared__ float rowmin[F_TILE_ROWS];

    const int tid     = threadIdx.x;
    const int lane    = tid & 31;
    const int wid     = tid >> 5;
    const int rowBase = blockIdx.x * F_TILE_ROWS;
    const int hi      = (lane >> 4) & 1;
    const int arow    = lane & 15;

    for (int i = tid; i < DIMIN; i += 256) bdS[i] = bd[i];
    if (tid < F_TILE_ROWS) { cnt[tid] = 0u; rowmin[tid] = 0.0f; }
    {
        float* tv = &topV[0][0];
        int*   ti = &topI[0][0];
        for (int i = tid; i < F_TILE_ROWS * TOPK; i += 256) { tv[i] = 0.0f; ti[i] = 0; }
    }
    for (int i = tid; i < F_TILE_ROWS * DIMIN; i += 256) {
        int r = i / DIMIN, c = i - r * DIMIN;
        xcb[i] = f32_to_bf16_rne(x[(size_t)(rowBase + r) * DIMIN + c] - bd[c]);
    }
    __syncthreads();

    const unsigned short* xrow = &xcb[arow * DIMIN];

    for (int ch = 0; ch < F_NCHUNKS; ++ch) {
        const int colBase0 = ch * F_CHUNK + wid * 16;
        const int colBase1 = colBase0 + 128;
        const float* B0f = &Ae[(size_t)(colBase0 + arow) * DIMIN];
        const float* B1f = &Ae[(size_t)(colBase1 + arow) * DIMIN];

        v8f acc0 = {0.f, 0.f, 0.f, 0.f, 0.f, 0.f, 0.f, 0.f};
        v8f acc1 = {0.f, 0.f, 0.f, 0.f, 0.f, 0.f, 0.f, 0.f};

        #pragma unroll 2
        for (int kk = 0; kk < DIMIN; kk += 32) {
            FragBF a, b0, b1;
            load_frag_a(a, xrow, kk, hi);
            const int ob = kk + hi * 16;           // B: 16 contiguous K per lane
            #pragma unroll
            for (int i = 0; i < 16; ++i) {
                b0.h[i] = f32_to_bf16_rne(B0f[ob + i]);
                b1.h[i] = f32_to_bf16_rne(B1f[ob + i]);
            }
            acc0 = __builtin_amdgcn_wmma_f32_16x16x32_bf16(
                       false, a.v, false, b0.v, (short)0, acc0, false, false);
            acc1 = __builtin_amdgcn_wmma_f32_16x16x32_bf16(
                       false, a.v, false, b1.v, (short)0, acc1, false, false);
        }

        const int c0 = colBase0 + (lane & 15);
        const int c1 = colBase1 + (lane & 15);
        #pragma unroll
        for (int r = 0; r < 8; ++r) {
            const int m = r + (hi << 3);
            float v0 = acc0[r];
            if (v0 > 0.0f && v0 > rowmin[m]) {
                unsigned p = atomicAdd(&cnt[m], 1u);
                if (p < F_CAP) { candV[m][p] = v0; candI[m][p] = (unsigned short)c0; }
            }
            float v1 = acc1[r];
            if (v1 > 0.0f && v1 > rowmin[m]) {
                unsigned p = atomicAdd(&cnt[m], 1u);
                if (p < F_CAP) { candV[m][p] = v1; candI[m][p] = (unsigned short)c1; }
            }
        }
        __syncthreads();

        if (tid < F_TILE_ROWS) {
            const int m = tid;
            unsigned n = cnt[m]; if (n > F_CAP) n = F_CAP;
            for (unsigned i = 0; i < n; ++i) {
                float v = candV[m][i];
                if (v > topV[m][TOPK - 1]) {
                    int j = TOPK - 1;
                    while (j > 0 && topV[m][j - 1] < v) {
                        topV[m][j] = topV[m][j - 1];
                        topI[m][j] = topI[m][j - 1];
                        --j;
                    }
                    topV[m][j] = v;
                    topI[m][j] = (int)candI[m][i];
                }
            }
            rowmin[m] = topV[m][TOPK - 1];
            cnt[m] = 0u;
        }
        __syncthreads();
    }

    float lam;
    {
        float lp = lambda_pre[0];
        lam = (lp > 20.0f) ? lp : log1pf(expf(lp));
    }
    for (int r = 0; r < F_TILE_ROWS; ++r) {
        for (int cb = 0; cb < DIMIN; cb += 256) {
            const int c = cb + tid;
            float acc = bdS[c];
            #pragma unroll 8
            for (int k = 0; k < TOPK; ++k) {
                acc += lam * topV[r][k] * Ae[(size_t)topI[r][k] * DIMIN + c];
            }
            out[(size_t)(rowBase + r) * DIMIN + c] = acc;
        }
    }
}

// ---------------------------------------------------------------------------
extern "C" void kernel_launch(void* const* d_in, const int* in_sizes, int n_in,
                              void* d_out, int out_size, void* d_ws, size_t ws_size,
                              hipStream_t stream) {
    (void)in_sizes; (void)n_in; (void)out_size;
    const float* x  = (const float*)d_in[0];
    const float* Ae = (const float*)d_in[1];
    // d_in[2] = Ad (== Ae^T, unused: we gather rows of Ae directly)
    const float* bd = (const float*)d_in[3];
    const float* lp = (const float*)d_in[4];
    // d_in[5] = kval (hardcoded TOPK=32 per reference setup)
    float* out = (float*)d_out;

    const size_t aebBytes = (size_t)WIDTH * DIMIN * sizeof(unsigned short);
    const size_t xcbBytes = (size_t)NROWS * DIMIN * sizeof(unsigned short);

    if (ws_size >= aebBytes + xcbBytes) {
        unsigned short* Aeb = (unsigned short*)d_ws;
        unsigned short* Xcb = (unsigned short*)((char*)d_ws + aebBytes);
        cvt_bf16_kernel<<<2048, 256, 0, stream>>>(Ae, Aeb, WIDTH * DIMIN);
        prep_xc_kernel<<<2048, 256, 0, stream>>>(x, bd, Xcb, NROWS * DIMIN);
        sae_main_ws<<<NROWS / W_TILE_ROWS, 256, 0, stream>>>(
            Ae, Aeb, Xcb, bd, lp, out);
    } else {
        sae_main_nows<<<NROWS / F_TILE_ROWS, 256, 0, stream>>>(
            x, Ae, bd, lp, out);
    }
}
